// Net_81698867904995
// MI455X (gfx1250) — 8-bit matrix operand order repaired and checked by reading, compiled, not run on hardware
//
#include <hip/hip_runtime.h>
#include <hip/hip_bf16.h>

typedef __attribute__((ext_vector_type(8))) int v8i;

#define WAVES_PER_BLOCK 8
#define DV_SCALE 0.08f
#define SWIZZLE_SWAPX16 0x401F

__global__ void snn_prep_params(const float* __restrict__ w,
                                const float* __restrict__ vth,
                                int O, int C,
                                float* __restrict__ alpha_out,
                                float* __restrict__ vthq_out,
                                signed char* __restrict__ wt_out)
{
    int o = threadIdx.x;
    if (o >= 16) return;
    int K = C * 16;
    if (o < O) {
        const float* wo = w + (size_t)o * K;
        float s_abs = 0.f;
        for (int k = 0; k < K; ++k) s_abs += fabsf(wo[k]);
        float thr = (s_abs / (float)K) * 0.7f;
        float s = 0.f, n = 0.f;
        for (int k = 0; k < K; ++k) {
            float a = fabsf(wo[k]);
            if (a > thr) { s += a; n += 1.f; }
        }
        float alpha = s / n;
        alpha_out[o] = alpha;
        float q = rintf(vth[o] / alpha);
        q = fminf(fmaxf(q, 8.f), 32.f);
        vthq_out[o] = q * alpha;
        for (int k = 0; k < K; ++k) {
            float wv = wo[k];
            float s1 = (wv + thr > 0.f) ? 1.f : ((wv + thr < 0.f) ? -1.f : 0.f);
            float s2 = (wv - thr > 0.f) ? 1.f : ((wv - thr < 0.f) ? -1.f : 0.f);
            float sg = s1 + s2;
            wt_out[(size_t)o * K + k] = (sg > 0.f) ? 1 : ((sg < 0.f) ? -1 : 0);
        }
    } else {
        alpha_out[o] = 1.f;
        vthq_out[o] = 1e30f;
    }
}

__global__ void snn_prep_bfrag(const signed char* __restrict__ wt,
                               int O, int C, int NC,
                               unsigned* __restrict__ bfrag)
{
    int gid = blockIdx.x * blockDim.x + threadIdx.x;
    int total = 3 * NC * 8 * 32;
    if (gid >= total) return;
    int lane = gid & 31;
    int v    = (gid >> 5) & 7;
    int c    = (gid >> 8) % NC;
    int m    = gid / (256 * NC);
    int N  = lane & 15;
    int hi = lane >> 4;
    int K  = C * 16;
    unsigned dw = 0;
    for (int byte = 0; byte < 4; ++byte) {
        int k = 64 * c + 32 * (v >> 2) + 16 * ((v >> 1) & 1) + 4 * (2 * hi + (v & 1)) + byte;
        unsigned val = 0;
        if (k < K && N < O) {
            signed char t = wt[(size_t)N * K + k];
            if (m == 0)      val = (t == 1)  ? 1u : 0u;
            else if (m == 1) val = (t == -1) ? 1u : 0u;
            else             val = (t == 0)  ? 1u : 0u;
        }
        dw |= val << (8 * byte);
    }
    bfrag[gid] = dw;
}

__global__ void snn_f32_to_u8(const float* __restrict__ x,
                              unsigned char* __restrict__ y, int n)
{
    int i = blockIdx.x * blockDim.x + threadIdx.x;
    if (i < n) y[i] = (x[i] >= 0.5f) ? 1 : 0;
}

template<int NC, int C, int O, int HIN, int HOUT>
__global__ __launch_bounds__(256, 1)
void snn_conv_lif(const unsigned char* __restrict__ in_spk,
                  unsigned char* __restrict__ out_spk,
                  const float* __restrict__ dv_table,
                  const float* __restrict__ alpha_v,
                  const float* __restrict__ vthq_v,
                  const unsigned* __restrict__ bfrag,
                  int total_tiles)
{
    constexpr int WIN = HIN, WOUT = HOUT;
    constexpr int NTX = (WOUT + 1) / 2;
    constexpr size_t IMG = (size_t)C * HIN * WIN;
    constexpr size_t HW  = (size_t)HOUT * WOUT;

    const int lane    = (int)threadIdx.x & 31;
    const int wave    = (int)threadIdx.x >> 5;
    const int wave_id = (int)blockIdx.x * WAVES_PER_BLOCK + wave;
    const int nwaves  = (int)gridDim.x * WAVES_PER_BLOCK;

    const int hi   = lane >> 4;
    const int nlo  = lane & 15;
    const int mrow = lane & 15;
    const int t_a  = mrow >> 1;
    const int xi_a = mrow & 1;

    v8i bfr[3][NC];
    #pragma unroll
    for (int m = 0; m < 3; ++m)
        #pragma unroll
        for (int c = 0; c < NC; ++c)
            #pragma unroll
            for (int v = 0; v < 8; ++v)
                bfr[m][c][v] = (int)bfrag[(size_t)(((m * NC) + c) * 8 + v) * 32 + lane];

    const float al       = alpha_v[nlo];
    const float sc       = al * DV_SCALE;
    const float vth      = vthq_v[nlo];
    const float vclip_lo = -32.0f * al;
    const float vclip_hi =  32.0f * al;

    float  pown[8];
    size_t pob    = 0;
    bool   pstore = false;
    bool   pvalid = false;
    #pragma unroll
    for (int d = 0; d < 8; ++d) pown[d] = 0.f;

    for (int tile = wave_id; ; tile += nwaves) {
        const bool cvalid = (tile < total_tiles);

        float  cown[8];
        size_t cob    = 0;
        bool   cstore = false;

        if (cvalid) {
            const int tx = tile % NTX;
            const int y  = (tile / NTX) % HOUT;
            const int b  = tile / (NTX * HOUT);
            const int x0 = tx * 2;
            int xA = x0 + xi_a;
            if (xA > WOUT - 1) xA = WOUT - 1;

            const int xo = x0 + hi;
            cstore = (nlo < O) && (xo < WOUT);
            cob    = ((size_t)(b * 8) * O + nlo) * HW + (size_t)y * WOUT + xo;

            const unsigned char* base = in_spk + (size_t)(b * 8 + t_a) * IMG
                                      + (size_t)(y + 2 * hi) * WIN + (size_t)xA;

            v8i acc0 = {}; v8i acc1 = {}; v8i acc2 = {};

            #pragma unroll
            for (int c = 0; c < NC; ++c) {
                v8i afrag;
                #pragma unroll
                for (int v = 0; v < 8; ++v) {
                    const int kb   = 64 * c + 32 * (v >> 2) + 16 * ((v >> 1) & 1) + 4 * (v & 1);
                    const int cidx = kb >> 4;
                    const int kyl  = (kb >> 2) & 3;
                    unsigned d = 0;
                    if (cidx < C) {
                        __builtin_memcpy(&d, base + (size_t)cidx * (HIN * WIN)
                                                + (size_t)kyl * WIN, 4);
                    }
                    afrag[v] = (int)d;
                }
                acc0 = __builtin_amdgcn_wmma_i32_16x16x64_iu8(false, afrag, false, bfr[0][c], acc0, false, false);
                acc1 = __builtin_amdgcn_wmma_i32_16x16x64_iu8(false, afrag, false, bfr[1][c], acc1, false, false);
                acc2 = __builtin_amdgcn_wmma_i32_16x16x64_iu8(false, afrag, false, bfr[2][c], acc2, false, false);
            }

            unsigned p01[8];
            #pragma unroll
            for (int d = 0; d < 8; ++d)
                p01[d] = (unsigned)acc0[d] * 257u + (unsigned)acc1[d];

            #pragma unroll
            for (int d = 0; d < 8; ++d) {
                unsigned idx = p01[d] * 257u + (unsigned)acc2[d];
                cown[d] = dv_table[idx];
            }
        }

        if (pvalid) {
            float par[8];
            #pragma unroll
            for (int d = 0; d < 8; ++d)
                par[d] = __int_as_float(
                    __builtin_amdgcn_ds_swizzle(__float_as_int(pown[d]),
                                                SWIZZLE_SWAPX16));

            float v_mem = 0.0f;
            #pragma unroll
            for (int t = 0; t < 8; ++t) {
                const int dd = 2 * (t & 3);
                float raw;
                if (t < 4) raw = hi ? par[dd + 1] : pown[dd];
                else       raw = hi ? pown[dd + 1] : par[dd];
                v_mem += raw * sc;
                float vq = fminf(fmaxf(v_mem, vclip_lo), vclip_hi);
                bool fire = (vq >= vth);
                if (pstore)
                    out_spk[pob + (size_t)t * O * HW] = fire ? 1 : 0;
                if (fire) v_mem = 0.0f;
            }
        }

        if (!cvalid) break;

        #pragma unroll
        for (int d = 0; d < 8; ++d) pown[d] = cown[d];
        pob    = cob;
        pstore = cstore;
        pvalid = true;
    }
}

__global__ void snn_mean(const unsigned char* __restrict__ spk,
                         float* __restrict__ out, int HW)
{
    __shared__ float red[256];
    const int p = (int)blockIdx.x;
    const unsigned char* s = spk + (size_t)p * HW;
    float acc = 0.f;
    for (int i = (int)threadIdx.x; i < HW; i += 256) acc += (float)s[i];
    red[threadIdx.x] = acc;
    __syncthreads();
    for (int st = 128; st > 0; st >>= 1) {
        if ((int)threadIdx.x < st) red[threadIdx.x] += red[threadIdx.x + st];
        __syncthreads();
    }
    if (threadIdx.x == 0) out[p] = red[0] / (float)HW;
}

extern "C" void kernel_launch(void* const* d_in, const int* in_sizes, int n_in,
                              void* d_out, int out_size, void* d_ws, size_t ws_size,
                              hipStream_t stream)
{
    const float* x      = (const float*)d_in[0];
    const float* w[5]   = { (const float*)d_in[1], (const float*)d_in[3],
                            (const float*)d_in[5], (const float*)d_in[7],
                            (const float*)d_in[9] };
    const float* vth[5] = { (const float*)d_in[2], (const float*)d_in[4],
                            (const float*)d_in[6], (const float*)d_in[8],
                            (const float*)d_in[10] };
    const float* dv_table = (const float*)d_in[11];

    unsigned char* ws8  = (unsigned char*)d_ws;
    unsigned char* spkA = ws8;
    unsigned char* spkB = ws8 + (16u << 20);
    unsigned char* pb   = ws8 + (32u << 20);

    struct LayerCfg { int C, Hin, Hout, O, NC; };
    const LayerCfg cfg[5] = {
        { 3, 64, 61, 16, 1 },
        {16, 61, 58, 16, 4 },
        {16, 58, 55, 16, 4 },
        {16, 55, 52, 16, 4 },
        {16, 52, 49,  6, 4 },
    };

    {
        int n = 32 * 8 * 3 * 64 * 64;
        snn_f32_to_u8<<<(n + 255) / 256, 256, 0, stream>>>(x, spkA, n);
    }

    float* alpha[5]; float* vthq[5]; unsigned* bfr[5];
    for (int l = 0; l < 5; ++l) {
        const LayerCfg& L = cfg[l];
        unsigned char* pl = pb + (size_t)l * 32768;
        alpha[l] = (float*)pl;
        vthq[l]  = alpha[l] + 16;
        signed char* wt = (signed char*)(pl + 128);
        bfr[l] = (unsigned*)(pl + 8192);

        snn_prep_params<<<1, 32, 0, stream>>>(w[l], vth[l], L.O, L.C,
                                              alpha[l], vthq[l], wt);
        int total = 3 * L.NC * 8 * 32;
        snn_prep_bfrag<<<(total + 255) / 256, 256, 0, stream>>>(
            wt, L.O, L.C, L.NC, bfr[l]);
    }

    const int GRID = 1024;

    auto tiles = [](int Hout) { return 32 * Hout * ((Hout + 1) / 2); };

    snn_conv_lif<1, 3, 16, 64, 61><<<GRID, 256, 0, stream>>>(
        spkA, spkB, dv_table, alpha[0], vthq[0], bfr[0], tiles(61));
    snn_conv_lif<4, 16, 16, 61, 58><<<GRID, 256, 0, stream>>>(
        spkB, spkA, dv_table, alpha[1], vthq[1], bfr[1], tiles(58));
    snn_conv_lif<4, 16, 16, 58, 55><<<GRID, 256, 0, stream>>>(
        spkA, spkB, dv_table, alpha[2], vthq[2], bfr[2], tiles(55));
    snn_conv_lif<4, 16, 16, 55, 52><<<GRID, 256, 0, stream>>>(
        spkB, spkA, dv_table, alpha[3], vthq[3], bfr[3], tiles(52));
    snn_conv_lif<4, 16, 6, 52, 49><<<GRID, 256, 0, stream>>>(
        spkA, spkB, dv_table, alpha[4], vthq[4], bfr[4], tiles(49));

    snn_mean<<<256 * 6, 256, 0, stream>>>(spkB, (float*)d_out, 49 * 49);
}
